// NodeEncoder_36197984370738
// MI455X (gfx1250) — compile-verified, matrix operand order checked by reading, not run on hardware
//
#include <hip/hip_runtime.h>
#include <math.h>

// ---------------------------------------------------------------------------
// Problem constants (match reference)
// ---------------------------------------------------------------------------
#define NN     50000          // nodes
#define EE     800000         // raw edges
#define ETOT   (EE + NN)      // + self loops
#define DIN    128
#define DD     128            // H*C
#define HH     4
#define CC     32
#define DOUT3  384            // fused xl|xr|res GEMM width
#define CT_N   (DOUT3 / 16)   // 24 col tiles
#define KT_N   (DIN / 32)     // 4 k tiles
#define BLK_ROWS 64           // row tiles per block (4 x 16)
#define NEG_SLOPE 0.2f
#define LN_EPS 1e-5f

typedef __attribute__((ext_vector_type(16))) __bf16 v16bf;
typedef __attribute__((ext_vector_type(8)))  __bf16 v8bf;
typedef __attribute__((ext_vector_type(8)))  float  v8f;

// ---------------------------------------------------------------------------
// fp32 -> bf16 elementwise (initial x conversion)
// ---------------------------------------------------------------------------
__global__ void k_to_bf16(const float* __restrict__ x, __bf16* __restrict__ y, int n) {
    int i = blockIdx.x * blockDim.x + threadIdx.x;
    if (i < n) y[i] = (__bf16)x[i];
}

// ---------------------------------------------------------------------------
// Pack Wl|Wr|Wres (each [128][128], row-major K x Nout) into the WMMA B-lane
// order: wpack[((ct*4 + kt)*32 + lane)*16 + slot]; also build biasCat[384].
// ---------------------------------------------------------------------------
__global__ void k_pack_w(const float* __restrict__ Wl, const float* __restrict__ Wr,
                         const float* __restrict__ Wres,
                         const float* __restrict__ bl, const float* __restrict__ br,
                         __bf16* __restrict__ wpack, float* __restrict__ biasCat) {
    int idx = blockIdx.x * blockDim.x + threadIdx.x;
    const int total = CT_N * KT_N * 32 * 16;
    if (idx < total) {
        int slot = idx & 15;
        int lane = (idx >> 4) & 31;
        int kt   = (idx >> 9) & (KT_N - 1);
        int ct   = idx >> 11;
        int half = lane >> 4;
        int n    = ct * 16 + (lane & 15);
        int k    = kt * 32 + ((slot < 8) ? (half * 8 + slot)
                                         : (16 + half * 8 + (slot - 8)));
        const float* W; int nc = n;
        if (n < 128)      { W = Wl; }
        else if (n < 256) { W = Wr;   nc = n - 128; }
        else              { W = Wres; nc = n - 256; }
        wpack[idx] = (__bf16)W[k * DD + nc];
    }
    if (idx < DOUT3) {
        biasCat[idx] = (idx < 128) ? bl[idx] : ((idx < 256) ? br[idx - 128] : 0.0f);
    }
}

// ---------------------------------------------------------------------------
// Register-blocked WMMA GEMM: out[N][384] = xbf[N][128] @ Wcat[128][384] + bias
// Block = 256 threads = 8 waves, computes 64 rows x 128 cols.
//   - A tile (64x128 bf16 = 16KB) staged to LDS via global_load_async_to_lds
//     (ASYNCcnt path), then read as WMMA A fragments with ds_load_b128.
//   - Wave w handles col tile ct = blockIdx.y*8 + w across 4 row tiles:
//     each B fragment is reused 4x from registers -> 16 WMMAs per wave.
// A 16x32 bf16 lane layout (ISA 7.12.2): lane l<16: M=l, K in {0..7,16..23};
// lane l>=16: M=l-16, K in {8..15,24..31}.
// ---------------------------------------------------------------------------
__global__ void __launch_bounds__(256)
k_gemm_wmma(const __bf16* __restrict__ xbf, const __bf16* __restrict__ wpack,
            const float* __restrict__ biasCat, float* __restrict__ out, int rows) {
    __shared__ __bf16 sA[BLK_ROWS * DIN];          // 16 KB

    const int tid  = threadIdx.x;
    const int wave = tid >> 5;
    const int lane = tid & 31;
    const int half = lane >> 4;
    const int l15  = lane & 15;
    const int m0   = blockIdx.x * BLK_ROWS;
    const int ct   = blockIdx.y * 8 + wave;

    // ---- async-stage A tile: 256 threads x 4 x 16B = 16 KB ----
    {
        const size_t gbase = (size_t)m0 * DIN;            // element offset
        const size_t gmax  = (size_t)rows * DIN - 8;      // clamp for tail block
#pragma unroll
        for (int i = 0; i < 4; ++i) {
            const int elem = (tid + i * 256) * 8;         // 8 bf16 per 16B chunk
            size_t ge = gbase + (size_t)elem;
            if (ge > gmax) ge = gmax;                     // tail clamp (stores guarded)
            unsigned lofs = (unsigned)(size_t)(const void*)(sA + elem);
            unsigned long long ga = (unsigned long long)(size_t)(const void*)(xbf + ge);
            asm volatile("global_load_async_to_lds_b128 %0, %1, off"
                         :: "v"(lofs), "v"(ga) : "memory");
        }
        asm volatile("s_wait_asynccnt 0x0" ::: "memory");
        __syncthreads();
    }

    // ---- 16 WMMAs: 4 row tiles x 4 k tiles, B reused from registers ----
    v8f acc[4] = {v8f{}, v8f{}, v8f{}, v8f{}};
    const __bf16* bptr = wpack + ((size_t)(ct * KT_N) * 32 + lane) * 16;
#pragma unroll
    for (int kt = 0; kt < KT_N; ++kt) {
        const v16bf b = *(const v16bf*)(bptr + (size_t)kt * (32 * 16));
        const int k0 = kt * 32;
#pragma unroll
        for (int rt = 0; rt < 4; ++rt) {
            const __bf16* ap = sA + (rt * 16 + l15) * DIN;
            v8bf lo = *(const v8bf*)(ap + k0 + half * 8);
            v8bf hi = *(const v8bf*)(ap + k0 + 16 + half * 8);
            v16bf a = __builtin_shufflevector(lo, hi, 0, 1, 2, 3, 4, 5, 6, 7,
                                                      8, 9, 10, 11, 12, 13, 14, 15);
            acc[rt] = __builtin_amdgcn_wmma_f32_16x16x32_bf16(
                          /*neg_a=*/false, a, /*neg_b=*/false, b,
                          /*c_mod=*/(short)0, acc[rt],
                          /*reuse_a=*/false, /*reuse_b=*/false);
        }
    }

    // ---- epilogue: bias add + guarded stores ----
    const int col = ct * 16 + l15;
    const float bv = biasCat[col];
#pragma unroll
    for (int rt = 0; rt < 4; ++rt) {
        const int rbase = m0 + rt * 16 + half * 8;
#pragma unroll
        for (int r = 0; r < 8; ++r) {
            const int row = rbase + r;
            if (row < rows)
                out[(size_t)row * DOUT3 + col] = acc[rt][r] + bv;
        }
    }
}

// ---------------------------------------------------------------------------
// smax = -inf, ssum = 0, agg = 0
// ---------------------------------------------------------------------------
__global__ void k_init(float* __restrict__ smax, float* __restrict__ ssum,
                       float* __restrict__ agg) {
    int i = blockIdx.x * blockDim.x + threadIdx.x;
    if (i < NN * HH) { smax[i] = -INFINITY; ssum[i] = 0.0f; }
    if (i < NN * DD) agg[i] = 0.0f;
}

__device__ inline void atomicMaxF(float* addr, float val) {
    int* ia = (int*)addr;
    int cur = __float_as_int(*addr);
    while (__int_as_float(cur) < val) {
        int assumed = cur;
        cur = atomicCAS(ia, assumed, __float_as_int(val));
        if (cur == assumed) break;
    }
}

// ---------------------------------------------------------------------------
// Pass 1: one wave per edge. score[e][h] = att[h] . leaky(xl[src] + xr[dst]);
// segment max via atomic CAS-max on smax[dst][h].
// gout rows: [xl(0..127) | xr(128..255) | res(256..383)], stride 384.
// ---------------------------------------------------------------------------
__global__ void k_edge_score(const float* __restrict__ gout,
                             const int* __restrict__ src, const int* __restrict__ dst,
                             const float* __restrict__ att,
                             float* __restrict__ score, float* __restrict__ smax) {
    int e = (blockIdx.x * blockDim.x + threadIdx.x) >> 5;
    int lane = threadIdx.x & 31;
    if (e >= ETOT) return;
    int s, d;
    if (e < EE) { s = src[e]; d = dst[e]; } else { s = d = e - EE; }

    const int c0 = lane * 4;
    const float4 a = *(const float4*)(gout + (size_t)s * DOUT3 + c0);
    const float4 b = *(const float4*)(gout + (size_t)d * DOUT3 + 128 + c0);
    const float4 w = *(const float4*)(att + c0);   // att flat [H*C] == channel idx

    float acc = 0.0f;
    float v;
    v = a.x + b.x; v = (v > 0.f) ? v : NEG_SLOPE * v; acc += v * w.x;
    v = a.y + b.y; v = (v > 0.f) ? v : NEG_SLOPE * v; acc += v * w.y;
    v = a.z + b.z; v = (v > 0.f) ? v : NEG_SLOPE * v; acc += v * w.z;
    v = a.w + b.w; v = (v > 0.f) ? v : NEG_SLOPE * v; acc += v * w.w;

    // reduce within each group of 8 lanes (one head = 32 channels)
#pragma unroll
    for (int m = 1; m < 8; m <<= 1) acc += __shfl_xor(acc, m, 32);

    if ((lane & 7) == 0) {
        int h = lane >> 3;
        score[(size_t)e * HH + h] = acc;
        atomicMaxF(&smax[d * HH + h], acc);
    }
}

// ---------------------------------------------------------------------------
// Pass 2: ex = exp(score - smax[dst]); score <- ex; ssum[dst] += ex
// ---------------------------------------------------------------------------
__global__ void k_edge_norm(float* __restrict__ score, const float* __restrict__ smax,
                            float* __restrict__ ssum, const int* __restrict__ dst) {
    int i = blockIdx.x * blockDim.x + threadIdx.x;
    if (i >= ETOT * HH) return;
    int e = i >> 2, h = i & 3;
    int d = (e < EE) ? dst[e] : (e - EE);
    float ex = __expf(score[i] - smax[d * HH + h]);
    score[i] = ex;
    atomicAdd(&ssum[d * HH + h], ex);
}

// ---------------------------------------------------------------------------
// Pass 3: one wave per edge. agg[dst] += alpha * xl[src]
// ---------------------------------------------------------------------------
__global__ void k_edge_scatter(const float* __restrict__ gout,
                               const int* __restrict__ src, const int* __restrict__ dst,
                               const float* __restrict__ score,
                               const float* __restrict__ ssum,
                               float* __restrict__ agg) {
    int e = (blockIdx.x * blockDim.x + threadIdx.x) >> 5;
    int lane = threadIdx.x & 31;
    if (e >= ETOT) return;
    int s, d;
    if (e < EE) { s = src[e]; d = dst[e]; } else { s = d = e - EE; }

    const int c0 = lane * 4;
    const int h = lane >> 3;
    const float alpha = score[(size_t)e * HH + h] / ssum[d * HH + h];
    const float4 a = *(const float4*)(gout + (size_t)s * DOUT3 + c0);
    float* p = agg + (size_t)d * DD + c0;
    atomicAdd(p + 0, a.x * alpha);
    atomicAdd(p + 1, a.y * alpha);
    atomicAdd(p + 2, a.z * alpha);
    atomicAdd(p + 3, a.w * alpha);
}

// ---------------------------------------------------------------------------
// Finalize: one wave per node. v = agg + res + bias; LayerNorm; ReLU.
// Writes fp32 output and bf16 copy (next layer's WMMA input).
// ---------------------------------------------------------------------------
__global__ void k_finalize(const float* __restrict__ agg, const float* __restrict__ gout,
                           const float* __restrict__ bias, const float* __restrict__ g,
                           const float* __restrict__ be,
                           float* __restrict__ out, __bf16* __restrict__ outbf) {
    int n = (blockIdx.x * blockDim.x + threadIdx.x) >> 5;
    int lane = threadIdx.x & 31;
    if (n >= NN) return;
    const int c0 = lane * 4;

    float v[4];
#pragma unroll
    for (int j = 0; j < 4; ++j)
        v[j] = agg[(size_t)n * DD + c0 + j]
             + gout[(size_t)n * DOUT3 + 256 + c0 + j]
             + bias[c0 + j];

    float tot = v[0] + v[1] + v[2] + v[3];
#pragma unroll
    for (int m = 1; m < 32; m <<= 1) tot += __shfl_xor(tot, m, 32);
    const float mean = tot * (1.0f / DD);

    float vs = 0.0f;
#pragma unroll
    for (int j = 0; j < 4; ++j) { float dv = v[j] - mean; vs += dv * dv; }
#pragma unroll
    for (int m = 1; m < 32; m <<= 1) vs += __shfl_xor(vs, m, 32);
    const float inv = rsqrtf(vs * (1.0f / DD) + LN_EPS);

#pragma unroll
    for (int j = 0; j < 4; ++j) {
        float o = (v[j] - mean) * inv * g[c0 + j] + be[c0 + j];
        o = fmaxf(o, 0.0f);
        out[(size_t)n * DD + c0 + j] = o;
        outbf[(size_t)n * DD + c0 + j] = (__bf16)o;
    }
}

// ---------------------------------------------------------------------------
// Host launcher
// ---------------------------------------------------------------------------
extern "C" void kernel_launch(void* const* d_in, const int* in_sizes, int n_in,
                              void* d_out, int out_size, void* d_ws, size_t ws_size,
                              hipStream_t stream) {
    (void)in_sizes; (void)n_in; (void)out_size; (void)ws_size;

    const float* x   = (const float*)d_in[0];
    const int*   ei  = (const int*)d_in[1];
    const int*   src = ei;
    const int*   dst = ei + EE;

    // per-layer params: Wl, bl, Wr, br, att, Wres, bias, g, be
    const float* P[2][9];
    for (int l = 0; l < 2; ++l)
        for (int j = 0; j < 9; ++j)
            P[l][j] = (const float*)d_in[2 + l * 9 + j];

    // workspace carve-up (256B aligned)
    size_t off = 0;
    auto carve = [&](size_t bytes) {
        void* p = (char*)d_ws + off;
        off += (bytes + 255) & ~(size_t)255;
        return p;
    };
    __bf16* xbf    = (__bf16*)carve((size_t)NN * DIN * 2);
    __bf16* wpack  = (__bf16*)carve((size_t)CT_N * KT_N * 32 * 16 * 2);
    float*  biasC  = (float*)carve(DOUT3 * 4);
    float*  gout   = (float*)carve((size_t)NN * DOUT3 * 4);
    float*  score  = (float*)carve((size_t)ETOT * HH * 4);
    float*  smax   = (float*)carve((size_t)NN * HH * 4);
    float*  ssum   = (float*)carve((size_t)NN * HH * 4);
    float*  agg    = (float*)carve((size_t)NN * DD * 4);
    float*  h0     = (float*)carve((size_t)NN * DD * 4);

    const int T = 256;
    // x -> bf16
    k_to_bf16<<<(NN * DIN + T - 1) / T, T, 0, stream>>>(x, xbf, NN * DIN);

    for (int l = 0; l < 2; ++l) {
        const float* Wl = P[l][0], *bl = P[l][1], *Wr = P[l][2], *br = P[l][3];
        const float* att = P[l][4], *Wres = P[l][5], *bias = P[l][6];
        const float* g = P[l][7], *be = P[l][8];
        float* out_f = (l == 0) ? h0 : (float*)d_out;

        k_pack_w<<<(CT_N * KT_N * 32 * 16 + T - 1) / T, T, 0, stream>>>(
            Wl, Wr, Wres, bl, br, wpack, biasC);

        dim3 gg((NN + BLK_ROWS - 1) / BLK_ROWS, DOUT3 / 128);   // 782 x 3
        k_gemm_wmma<<<gg, T, 0, stream>>>(xbf, wpack, biasC, gout, NN);

        k_init<<<(NN * DD + T - 1) / T, T, 0, stream>>>(smax, ssum, agg);

        k_edge_score<<<((size_t)ETOT * 32 + T - 1) / T, T, 0, stream>>>(
            gout, src, dst, att, score, smax);
        k_edge_norm<<<((size_t)ETOT * HH + T - 1) / T, T, 0, stream>>>(
            score, smax, ssum, dst);
        k_edge_scatter<<<((size_t)ETOT * 32 + T - 1) / T, T, 0, stream>>>(
            gout, src, dst, score, ssum, agg);

        k_finalize<<<((size_t)NN * 32 + T - 1) / T, T, 0, stream>>>(
            agg, gout, bias, g, be, out_f, xbf);
    }
}
